// GEARSageSafe_7490422964617
// MI455X (gfx1250) — compile-verified, matrix operand order checked by reading, not run on hardware
//
#include <hip/hip_runtime.h>
#include <hip/hip_bf16.h>

#define NODES 50000
#define NEDGE 800000
#define EPSB 1e-5f

typedef __attribute__((ext_vector_type(16))) __bf16 v16bf;
typedef __attribute__((ext_vector_type(8)))  float  v8f;
typedef __attribute__((ext_vector_type(4)))  unsigned int v4u;
typedef unsigned short ushort_t;

union Frag {
    v16bf   v;
    ushort_t u[16];
    v4u      q[2];
};

__device__ __forceinline__ ushort_t f2bf(float f) {
    unsigned int u = __float_as_uint(f);
    unsigned int r = u + 0x7FFFu + ((u >> 16) & 1u);   // round-to-nearest-even
    return (ushort_t)(r >> 16);
}

// ---------------- generic zero ----------------
__global__ void k_zero_f32(float* p, int n) {
    int i = blockIdx.x * blockDim.x + threadIdx.x;
    if (i < n) p[i] = 0.0f;
}

// ---------------- edge features: aggE[row] += [ea ; et], padded to 128 cols ----------------
__global__ void k_edge_feat(float* __restrict__ aggE,
                            const int* __restrict__ ei, const int* __restrict__ attr,
                            const int* __restrict__ tt, const int* __restrict__ dd,
                            const float* __restrict__ emb_type, const float* __restrict__ emb_dir,
                            const float* __restrict__ w_t, const float* __restrict__ b_t) {
    int e = blockIdx.x;
    int j = threadIdx.x;
    if (j >= 100) return;
    int r = ei[e];                       // edge_index[0][e]
    float val;
    if (j < 50) {
        val = emb_type[attr[e] * 50 + j] + emb_dir[dd[e] * 50 + j];
    } else {
        int jj = j - 50;
        float tl = log1pf((float)tt[e]);
        val = cosf(tl * w_t[jj] + b_t[jj]);
    }
    atomicAdd(&aggE[r * 128 + j], val);
}

// ---------------- scatter: aggH[row] += h[col] (128 floats / edge, 8 edges per block) ----------------
__global__ void k_scatter_h(float* __restrict__ aggH, const float* __restrict__ h,
                            const int* __restrict__ ei) {
    int e = blockIdx.x * 8 + (threadIdx.x >> 5);
    int lane = threadIdx.x & 31;
    int r = ei[e];
    int c = ei[NEDGE + e];
    const float4 v = *(const float4*)(h + c * 128 + lane * 4);
    float* dst = aggH + r * 128 + lane * 4;
    atomicAdd(dst + 0, v.x);
    atomicAdd(dst + 1, v.y);
    atomicAdd(dst + 2, v.z);
    atomicAdd(dst + 3, v.w);
}

// ---------------- f32 [N,128] -> bf16 swizzled into WMMA A-fragment layout ----------------
// dst index = ((rt*4 + kt)*32 + lane)*16 + j ; per ISA 7.12.2 16-bit 16x32 A layout
__global__ void k_cvt_swz(ushort_t* __restrict__ dst, const float* __restrict__ src) {
    int tid = blockIdx.x * blockDim.x + threadIdx.x;
    if (tid >= (NODES / 16) * 2048) return;
    int j    = tid & 15;
    int lane = (tid >> 4) & 31;
    int kt   = (tid >> 9) & 3;
    int rt   = tid >> 11;
    int row = rt * 16 + (lane & 15);
    int k   = kt * 32 + ((j & 8) ? 16 : 0) + ((lane & 16) ? 8 : 0) + (j & 7);
    dst[tid] = f2bf(src[row * 128 + k]);
}

// ---------------- weight prep: split Wm, pad, transpose, swizzle into B-fragment layout ----------------
// B[k][n] = W[n][k]; dst index = ((kt*ntiles + nt)*32 + lane)*16 + j
__global__ void k_prep_w(const float* __restrict__ Wm, const float* __restrict__ Wr,
                         ushort_t* __restrict__ Bh, ushort_t* __restrict__ Be,
                         ushort_t* __restrict__ Br, int dout, int ntiles) {
    int tid = blockIdx.x * blockDim.x + threadIdx.x;
    int total = 4 * ntiles * 512;
    if (tid >= total) return;
    int j    = tid & 15;
    int lane = (tid >> 4) & 31;
    int rest = tid >> 9;
    int nt   = rest % ntiles;
    int kt   = rest / ntiles;
    int n = nt * 16 + (lane & 15);
    int k = kt * 32 + ((j & 8) ? 16 : 0) + ((lane & 16) ? 8 : 0) + (j & 7);
    const int fm = 128 + 100;
    float vh = (n < dout) ? Wm[n * fm + k] : 0.0f;
    float ve = (n < dout && k < 100) ? Wm[n * fm + 128 + k] : 0.0f;
    float vr = (n < dout) ? Wr[n * 128 + k] : 0.0f;
    Bh[tid] = f2bf(vh);
    Be[tid] = f2bf(ve);
    Br[tid] = f2bf(vr);
}

// ---------------- fused GEMM epilogue kernel ----------------
// out = 0.5*(aggH@WmH.T + aggE@WmE.T + bm) + h@Wr.T + br
// one wave per 16x16 output tile; blockDim = 32*ntiles; grid = NODES/16
__global__ void k_gemm_wmma(const ushort_t* __restrict__ Ah, const ushort_t* __restrict__ Ae,
                            const ushort_t* __restrict__ Ar,
                            const ushort_t* __restrict__ Bh, const ushort_t* __restrict__ Be,
                            const ushort_t* __restrict__ Br,
                            const float* __restrict__ bm, const float* __restrict__ brv,
                            float* __restrict__ out, int ldc, int dout, int ntiles) {
    const int lane = threadIdx.x & 31;
    const int nt   = threadIdx.x >> 5;
    const int rt   = blockIdx.x;

    v8f cM = {0.f, 0.f, 0.f, 0.f, 0.f, 0.f, 0.f, 0.f};
    v8f cR = {0.f, 0.f, 0.f, 0.f, 0.f, 0.f, 0.f, 0.f};
    Frag a, b;

    const int abase = (rt * 4) * 32 + lane;          // A fragment row-tile base
    __builtin_prefetch(Ae + (size_t)abase * 16, 0, 0);
    __builtin_prefetch(Ar + (size_t)abase * 16, 0, 0);

#pragma unroll
    for (int kt = 0; kt < 4; ++kt) {
        const v4u* pa = (const v4u*)(Ah + (size_t)((abase + kt * 32)) * 16);
        const v4u* pb = (const v4u*)(Bh + (size_t)(((kt * ntiles + nt) * 32 + lane)) * 16);
        a.q[0] = pa[0]; a.q[1] = pa[1];
        b.q[0] = pb[0]; b.q[1] = pb[1];
        cM = __builtin_amdgcn_wmma_f32_16x16x32_bf16(false, a.v, false, b.v, (short)0, cM, false, false);
    }
#pragma unroll
    for (int kt = 0; kt < 4; ++kt) {
        const v4u* pa = (const v4u*)(Ae + (size_t)((abase + kt * 32)) * 16);
        const v4u* pb = (const v4u*)(Be + (size_t)(((kt * ntiles + nt) * 32 + lane)) * 16);
        a.q[0] = pa[0]; a.q[1] = pa[1];
        b.q[0] = pb[0]; b.q[1] = pb[1];
        cM = __builtin_amdgcn_wmma_f32_16x16x32_bf16(false, a.v, false, b.v, (short)0, cM, false, false);
    }
#pragma unroll
    for (int kt = 0; kt < 4; ++kt) {
        const v4u* pa = (const v4u*)(Ar + (size_t)((abase + kt * 32)) * 16);
        const v4u* pb = (const v4u*)(Br + (size_t)(((kt * ntiles + nt) * 32 + lane)) * 16);
        a.q[0] = pa[0]; a.q[1] = pa[1];
        b.q[0] = pb[0]; b.q[1] = pb[1];
        cR = __builtin_amdgcn_wmma_f32_16x16x32_bf16(false, a.v, false, b.v, (short)0, cR, false, false);
    }

    int n = nt * 16 + (lane & 15);
    int mbase = rt * 16 + ((lane & 16) ? 8 : 0);
    if (n < dout) {
        float bias = 0.5f * bm[n] + brv[n];
#pragma unroll
        for (int r2 = 0; r2 < 8; ++r2) {
            out[(size_t)(mbase + r2) * ldc + n] = 0.5f * cM[r2] + cR[r2] + bias;
        }
    }
}

// ---------------- BN column stats: stats[j]=sum, stats[dout+j]=sumsq ----------------
__global__ void k_bn_stats(const float* __restrict__ out, float* __restrict__ stats,
                           int dout, int ldc) {
    int t = blockIdx.x * blockDim.x + threadIdx.x;
    int T = gridDim.x * blockDim.x;
    int j = t % dout;
    int stride = T / dout;
    float s = 0.f, ss = 0.f;
    for (int row = t / dout; row < NODES; row += stride) {
        float v = out[(size_t)row * ldc + j];
        s += v; ss += v * v;
    }
    atomicAdd(&stats[j], s);
    atomicAdd(&stats[dout + j], ss);
}

// ---------------- BN apply + ELU (dout==128 layers) ----------------
__global__ void k_bn_apply_elu(const float* __restrict__ out, const float* __restrict__ stats,
                               const float* __restrict__ g, const float* __restrict__ be,
                               float* __restrict__ h) {
    int idx = blockIdx.x * blockDim.x + threadIdx.x;
    if (idx >= NODES * 128) return;
    int j = idx & 127;
    const float invN = 1.0f / (float)NODES;
    float mu  = stats[j] * invN;
    float var = stats[128 + j] * invN - mu * mu;
    float v = out[idx];
    float y = (v - mu) * rsqrtf(var + EPSB) * g[j] + be[j];
    y = (y > 0.f) ? y : expm1f(y);          // ELU(alpha=1)
    h[idx] = y;
}

// ---------------- final: BN + log_softmax over 2 classes ----------------
__global__ void k_bn_logsoftmax(const float* __restrict__ out, const float* __restrict__ stats,
                                const float* __restrict__ g, const float* __restrict__ be,
                                float* __restrict__ o, int ldc) {
    int row = blockIdx.x * blockDim.x + threadIdx.x;
    if (row >= NODES) return;
    const float invN = 1.0f / (float)NODES;
    float y[2];
#pragma unroll
    for (int j = 0; j < 2; ++j) {
        float mu  = stats[j] * invN;
        float var = stats[2 + j] * invN - mu * mu;
        float v = out[(size_t)row * ldc + j];
        y[j] = (v - mu) * rsqrtf(var + EPSB) * g[j] + be[j];
    }
    float mx = fmaxf(y[0], y[1]);
    float lse = mx + logf(expf(y[0] - mx) + expf(y[1] - mx));
    o[row * 2 + 0] = y[0] - lse;
    o[row * 2 + 1] = y[1] - lse;
}

extern "C" void kernel_launch(void* const* d_in, const int* in_sizes, int n_in,
                              void* d_out, int out_size, void* d_ws, size_t ws_size,
                              hipStream_t stream) {
    (void)in_sizes; (void)n_in; (void)out_size; (void)ws_size;
    const float* x        = (const float*)d_in[0];
    const int*   ei       = (const int*)d_in[1];   // [2,E]
    const int*   eattr    = (const int*)d_in[2];
    const int*   etime    = (const int*)d_in[3];
    const int*   edir     = (const int*)d_in[4];
    const float* emb_type = (const float*)d_in[5];
    const float* emb_dir  = (const float*)d_in[6];
    const float* w_t      = (const float*)d_in[7];
    const float* b_t      = (const float*)d_in[8];
    const float* Wm[3]  = {(const float*)d_in[9],  (const float*)d_in[15], (const float*)d_in[21]};
    const float* bm[3]  = {(const float*)d_in[10], (const float*)d_in[16], (const float*)d_in[22]};
    const float* Wr[3]  = {(const float*)d_in[11], (const float*)d_in[17], (const float*)d_in[23]};
    const float* brv[3] = {(const float*)d_in[12], (const float*)d_in[18], (const float*)d_in[24]};
    const float* g[3]   = {(const float*)d_in[13], (const float*)d_in[19], (const float*)d_in[25]};
    const float* be[3]  = {(const float*)d_in[14], (const float*)d_in[20], (const float*)d_in[26]};

    // ---- workspace layout ----
    size_t off = 0;
    auto alloc = [&](size_t bytes) -> void* {
        void* p = (char*)d_ws + off;
        off += (bytes + 255) & ~(size_t)255;
        return p;
    };
    float*    aggE    = (float*)alloc((size_t)NODES * 128 * 4);
    float*    aggH    = (float*)alloc((size_t)NODES * 128 * 4);
    float*    outb    = aggH;   // alias: aggH(f32) dead once converted to bf16, GEMM reuses it
    float*    hbuf    = (float*)alloc((size_t)NODES * 128 * 4);
    ushort_t* aggE_bf = (ushort_t*)alloc((size_t)NODES * 128 * 2);
    ushort_t* aggH_bf = (ushort_t*)alloc((size_t)NODES * 128 * 2);
    ushort_t* h_bf    = (ushort_t*)alloc((size_t)NODES * 128 * 2);
    ushort_t* Bh      = (ushort_t*)alloc(4 * 8 * 512 * 2);
    ushort_t* Be      = (ushort_t*)alloc(4 * 8 * 512 * 2);
    ushort_t* Br      = (ushort_t*)alloc(4 * 8 * 512 * 2);
    float*    stats   = (float*)alloc(256 * 4);

    const int nElem = NODES * 128;
    const int swzThreads = (NODES / 16) * 2048;

    // ---- layer-invariant edge aggregation ----
    k_zero_f32<<<(nElem + 255) / 256, 256, 0, stream>>>(aggE, nElem);
    k_edge_feat<<<NEDGE, 128, 0, stream>>>(aggE, ei, eattr, etime, edir,
                                           emb_type, emb_dir, w_t, b_t);
    k_cvt_swz<<<(swzThreads + 255) / 256, 256, 0, stream>>>(aggE_bf, aggE);
    k_cvt_swz<<<(swzThreads + 255) / 256, 256, 0, stream>>>(h_bf, x);

    for (int li = 0; li < 3; ++li) {
        const int dout   = (li == 2) ? 2 : 128;
        const int ntiles = (li == 2) ? 1 : 8;
        const int ldc    = ntiles * 16;
        const float* h_cur = (li == 0) ? x : hbuf;

        k_prep_w<<<(4 * ntiles * 512 + 255) / 256, 256, 0, stream>>>(
            Wm[li], Wr[li], Bh, Be, Br, dout, ntiles);

        k_zero_f32<<<(nElem + 255) / 256, 256, 0, stream>>>(aggH, nElem);
        k_scatter_h<<<NEDGE / 8, 256, 0, stream>>>(aggH, h_cur, ei);
        k_cvt_swz<<<(swzThreads + 255) / 256, 256, 0, stream>>>(aggH_bf, aggH);

        k_gemm_wmma<<<NODES / 16, 32 * ntiles, 0, stream>>>(
            aggH_bf, aggE_bf, h_bf, Bh, Be, Br, bm[li], brv[li],
            outb, ldc, dout, ntiles);

        k_zero_f32<<<1, 256, 0, stream>>>(stats, 2 * dout);
        k_bn_stats<<<512, 256, 0, stream>>>(outb, stats, dout, ldc);

        if (li < 2) {
            k_bn_apply_elu<<<(nElem + 255) / 256, 256, 0, stream>>>(
                outb, stats, g[li], be[li], hbuf);
            k_cvt_swz<<<(swzThreads + 255) / 256, 256, 0, stream>>>(h_bf, hbuf);
        } else {
            k_bn_logsoftmax<<<(NODES + 255) / 256, 256, 0, stream>>>(
                outb, stats, g[li], be[li], (float*)d_out, ldc);
        }
    }
}
